// Attention_mca_aware_g2l_12103217840308
// MI455X (gfx1250) — compile-verified, matrix operand order checked by reading, not run on hardware
//
#include <hip/hip_runtime.h>

// ---------------------------------------------------------------------------
// MI455X (gfx1250, wave32) implementation of the MCA-aware G2L attention ref.
// All GEMMs go through v_wmma_f32_16x16x32_bf16 (bf16 in, f32 accum).
// Tile staging uses the CDNA5 direct global->LDS async path when available
// (GLOBAL_LOAD_ASYNC_TO_LDS_B128, ASYNCcnt), falling back to load+ds_store.
// Attention maps are kept in bf16 to halve HBM traffic (the dominant cost).
// Workspace requirement: ~360 MB.
// ---------------------------------------------------------------------------

typedef __attribute__((ext_vector_type(16))) __bf16 v16bf;
typedef __attribute__((ext_vector_type(8)))  __bf16 v8bf;
typedef __attribute__((ext_vector_type(8)))  float  v8f;
typedef __attribute__((ext_vector_type(4)))  int    v4i;

#if __has_builtin(__builtin_amdgcn_global_load_async_to_lds_b128)
#define HAVE_ASYNC_LDS 1
#else
#define HAVE_ASYNC_LDS 0
#endif

#if HAVE_ASYNC_LDS
typedef __attribute__((address_space(1))) v4i gas_v4i;  // global AS pointer target
typedef __attribute__((address_space(3))) v4i lds_v4i;  // LDS AS pointer target
#endif

#define N_TOK 2048
#define C_DIM 256
#define H_NUM 8
#define D_DIM 32
#define C2_DIM 512
#define SCALE_F 25.0f
#define SIM_TH 0.75f
#define CONF_TH 0.99f

// ---------------------------------------------------------------------------
// Fragment loader: 16 bf16 elements per lane from an LDS row.
// Layout per CDNA5 ISA (16-bit A 16x32): lane = {m = l&15, kh = l>>4};
// elems 0..7 -> K = kh*8 + j ; elems 8..15 -> K = 16 + kh*8 + (j-8).
// B (32x16) is staged transposed so it uses the identical pattern.
// ---------------------------------------------------------------------------
__device__ __forceinline__ v16bf ld_frag(const __bf16* p) {
    v8bf lo = *(const v8bf*)p;
    v8bf hi = *(const v8bf*)(p + 16);
    return __builtin_shufflevector(lo, hi, 0, 1, 2, 3, 4, 5, 6, 7,
                                   8, 9, 10, 11, 12, 13, 14, 15);
}

__device__ __forceinline__ void zero_row16(__bf16* p) {
    v8bf z;
#pragma unroll
    for (int e = 0; e < 8; ++e) z[e] = (__bf16)0.0f;
    *(v8bf*)p = z;
}

// Stage one 16-byte chunk global->LDS (async direct path if available).
__device__ __forceinline__ void stage_b128(const __bf16* gsrc, __bf16* ldst) {
#if HAVE_ASYNC_LDS
    __builtin_amdgcn_global_load_async_to_lds_b128((gas_v4i*)gsrc,
                                                   (lds_v4i*)ldst, 0, 0);
#else
    *(v8bf*)ldst = *(const v8bf*)gsrc;
#endif
}

__device__ __forceinline__ void stage_fence() {
#if HAVE_ASYNC_LDS
    asm volatile("s_wait_asynccnt 0" ::: "memory");
#endif
}

// ---------------------------------------------------------------------------
// Generic bf16 WMMA GEMM:  C[M,N] = alpha * A @ op(B) (+ bias[col])
//   NT=false: B is [K,N] row-major.  NT=true: B is [N,K] row-major (B^T).
// Tile: 128x64 per workgroup, BK=32, 8 waves each computing 32x32 (2x2 WMMA).
// Batch via grid.z with element strides sAz/sBz/sCz (supports per-head column
// slices: stride 32 columns).
// ---------------------------------------------------------------------------
template <bool NT>
__global__ __launch_bounds__(256) void gemm_bf16(
    const __bf16* __restrict__ A, const __bf16* __restrict__ B,
    float* __restrict__ C, const float* __restrict__ bias,
    int M, int N, int K, int lda, int ldb, int ldc,
    long long sAz, long long sBz, long long sCz, float alpha) {
    constexpr int BM = 128, BN = 64, BK = 32, LDW = BK + 8;  // 80B row: 16B aligned
    __shared__ __align__(16) __bf16 As[BM * LDW];
    __shared__ __align__(16) __bf16 Bs[BN * LDW];

    A += (long long)blockIdx.z * sAz;
    B += (long long)blockIdx.z * sBz;
    C += (long long)blockIdx.z * sCz;

    const int tile_n0 = blockIdx.x * BN;
    const int tile_m0 = blockIdx.y * BM;
    const int tid = threadIdx.x;
    const int lane = tid & 31;
    const int wid = tid >> 5;
    const int wr = wid >> 1;  // 0..3 : wave row (32 rows each)
    const int wc = wid & 1;   // 0..1 : wave col (32 cols each)

    v8f acc[2][2];
#pragma unroll
    for (int i = 0; i < 2; ++i)
#pragma unroll
        for (int j = 0; j < 2; ++j)
#pragma unroll
            for (int e = 0; e < 8; ++e) acc[i][j][e] = 0.0f;

    const int kh = lane >> 4;   // 0/1 : K-half selector
    const int rl = lane & 15;   // row/col within 16

    for (int k0 = 0; k0 < K; k0 += BK) {
        __syncthreads();
        // ---- stage A tile: 128x32, direct global->LDS b128 chunks ----------
        {
            const int c = tid & 3;       // 8-elem chunk within row
            const int r = tid >> 2;      // 0..63
#pragma unroll
            for (int p = 0; p < 2; ++p) {
                const int row = r + p * 64;
                const int gr = tile_m0 + row;
                __bf16* dst = As + row * LDW + c * 8;
                if (gr < M)
                    stage_b128(A + (long long)gr * lda + k0 + c * 8, dst);
                else
                    zero_row16(dst);
            }
        }
        // ---- stage B tile (transposed to [n][k] in LDS) --------------------
        if (NT) {
            const int c = tid & 3;
            const int r = tid >> 2;  // 0..63 == n
            const int gn = tile_n0 + r;
            __bf16* dst = Bs + r * LDW + c * 8;
            if (gn < N)
                stage_b128(B + (long long)gn * ldb + k0 + c * 8, dst);
            else
                zero_row16(dst);
        } else {
            if (k0 + BK < K)  // prefetch next k-slab of the streamed operand
                __builtin_prefetch(B + (long long)(k0 + BK) * ldb + tile_n0 + (tid & 63),
                                   0, 1);
#pragma unroll
            for (int i = 0; i < 8; ++i) {
                const int idx = i * 256 + tid;  // 0..2047
                const int kk = idx >> 6;        // 0..31
                const int nn = idx & 63;        // coalesced along N
                const int gn = tile_n0 + nn;
                float val = (gn < N) ? (float)B[(long long)(k0 + kk) * ldb + gn] : 0.0f;
                Bs[nn * LDW + kk] = (__bf16)val;
            }
        }
        stage_fence();
        __syncthreads();

        // ---- fragments + 4 WMMAs ------------------------------------------
        v16bf af[2], bf[2];
#pragma unroll
        for (int t = 0; t < 2; ++t) {
            af[t] = ld_frag(As + (wr * 32 + t * 16 + rl) * LDW + kh * 8);
            bf[t] = ld_frag(Bs + (wc * 32 + t * 16 + rl) * LDW + kh * 8);
        }
#pragma unroll
        for (int ti = 0; ti < 2; ++ti)
#pragma unroll
            for (int tj = 0; tj < 2; ++tj)
                acc[ti][tj] = __builtin_amdgcn_wmma_f32_16x16x32_bf16(
                    false, af[ti], false, bf[tj], (short)0, acc[ti][tj], false, false);
    }

    // ---- epilogue: C 16x16 f32 layout: row = vgpr + (lane>>4)*8, col = lane&15
    const int lh = lane >> 4;
#pragma unroll
    for (int ti = 0; ti < 2; ++ti)
#pragma unroll
        for (int tj = 0; tj < 2; ++tj) {
            const int col = tile_n0 + wc * 32 + tj * 16 + rl;
            if (col < N) {
                const float bv = bias ? bias[col] : 0.0f;
#pragma unroll
                for (int v = 0; v < 8; ++v) {
                    const int row = tile_m0 + wr * 32 + ti * 16 + lh * 8 + v;
                    if (row < M)
                        C[(long long)row * ldc + col] = alpha * acc[ti][tj][v] + bv;
                }
            }
        }
}

// ---------------------------------------------------------------------------
// Block reductions (256 threads, wave32)
// ---------------------------------------------------------------------------
__device__ __forceinline__ float blk_max(float v, float* red) {
#pragma unroll
    for (int m = 16; m >= 1; m >>= 1) v = fmaxf(v, __shfl_xor(v, m, 32));
    if ((threadIdx.x & 31) == 0) red[threadIdx.x >> 5] = v;
    __syncthreads();
    if (threadIdx.x == 0) {
        float r = red[0];
#pragma unroll
        for (int w = 1; w < 8; ++w) r = fmaxf(r, red[w]);
        red[0] = r;
    }
    __syncthreads();
    float out = red[0];
    __syncthreads();
    return out;
}

__device__ __forceinline__ float blk_sum(float v, float* red) {
#pragma unroll
    for (int m = 16; m >= 1; m >>= 1) v += __shfl_xor(v, m, 32);
    if ((threadIdx.x & 31) == 0) red[threadIdx.x >> 5] = v;
    __syncthreads();
    if (threadIdx.x == 0) {
        float r = red[0];
#pragma unroll
        for (int w = 1; w < 8; ++w) r += red[w];
        red[0] = r;
    }
    __syncthreads();
    float out = red[0];
    __syncthreads();
    return out;
}

// ---------------------------------------------------------------------------
// Row softmax with optional score/mask modulation (cls path).
// One block per (q,h) row; 2048 keys -> 8 values per thread in registers.
// ---------------------------------------------------------------------------
__global__ __launch_bounds__(256) void softmax_rows(
    const float* __restrict__ logits, __bf16* __restrict__ out,
    const float* __restrict__ score, int n, float scale) {
    __shared__ float red[8];
    const int q = blockIdx.x;
    const int h = blockIdx.y;
    const float* row = logits + ((long long)h * n + q) * n;
    __bf16* orow = out + ((long long)h * n + q) * n;
    const float sq = score ? (score[q] - 0.1f) : 0.0f;

    float loc[8];
    float mx = -3.0e38f;
#pragma unroll
    for (int i = 0; i < 8; ++i) {
        const int k = threadIdx.x + i * 256;
        float v = 0.0f;
        if (k < n) {
            v = row[k] * scale;
            if (score) {
                const float sk = score[k];
                v *= sk * ((sk > sq) ? 1.0f : 0.0f);
            }
        } else {
            v = -3.0e38f;
        }
        loc[i] = v;
        mx = fmaxf(mx, v);
    }
    mx = blk_max(mx, red);
    float sum = 0.0f;
#pragma unroll
    for (int i = 0; i < 8; ++i) {
        const int k = threadIdx.x + i * 256;
        float e = (k < n) ? __expf(loc[i] - mx) : 0.0f;
        loc[i] = e;
        sum += e;
    }
    sum = blk_sum(sum, red);
    const float inv = 1.0f / sum;
#pragma unroll
    for (int i = 0; i < 8; ++i) {
        const int k = threadIdx.x + i * 256;
        if (k < n) orow[k] = (__bf16)(loc[i] * inv);
    }
}

// ---------------------------------------------------------------------------
// Second-round similarity gating. One block per query row.
//   s       = mean_h attn_avg[h,q,:]            (softmax -> p)
//   sim2    = normalize( (acr>.75)*clsm*fgm * p )
//   obj     = normalize( (arr>.99) * sim2 )
// ---------------------------------------------------------------------------
__global__ __launch_bounds__(256) void sim_round2_kernel(
    const __bf16* __restrict__ attn_avg, const float* __restrict__ acr,
    const float* __restrict__ arr, const float* __restrict__ cls_score,
    const float* __restrict__ fg_score, __bf16* __restrict__ sim2,
    __bf16* __restrict__ obj, int n) {
    __shared__ float red[8];
    const int q = blockIdx.x;
    const float cq = cls_score[q] - 0.1f;
    const float fq = fg_score[q] - 0.1f;

    float loc[8];
    float mx = -3.0e38f;
#pragma unroll
    for (int i = 0; i < 8; ++i) {
        const int k = threadIdx.x + i * 256;
        float acc = 0.0f;
        if (k < n) {
#pragma unroll
            for (int h = 0; h < H_NUM; ++h)
                acc += (float)attn_avg[((long long)h * n + q) * n + k];
            acc *= (1.0f / H_NUM);
        } else {
            acc = -3.0e38f;
        }
        loc[i] = acc;
        mx = fmaxf(mx, acc);
    }
    mx = blk_max(mx, red);
    float sum = 0.0f;
#pragma unroll
    for (int i = 0; i < 8; ++i) {
        const int k = threadIdx.x + i * 256;
        float e = (k < n) ? __expf(loc[i] - mx) : 0.0f;
        loc[i] = e;
        sum += e;
    }
    sum = blk_sum(sum, red);
    const float inv = 1.0f / sum;

    float w1sum = 0.0f;
#pragma unroll
    for (int i = 0; i < 8; ++i) {
        const int k = threadIdx.x + i * 256;
        float w1 = 0.0f;
        if (k < n) {
            const float p = loc[i] * inv;
            const bool m = (acr[(long long)q * n + k] > SIM_TH) &&
                           (cls_score[k] > cq) && (fg_score[k] > fq);
            w1 = m ? p : 0.0f;
        }
        loc[i] = w1;
        w1sum += w1;
    }
    w1sum = blk_sum(w1sum, red);
    const float inv1 = 1.0f / w1sum;

    float w2sum = 0.0f;
#pragma unroll
    for (int i = 0; i < 8; ++i) {
        const int k = threadIdx.x + i * 256;
        if (k < n) {
            const float t1 = loc[i] * inv1;
            sim2[(long long)q * n + k] = (__bf16)t1;
            const float w2 = (arr[(long long)q * n + k] > CONF_TH) ? t1 : 0.0f;
            loc[i] = w2;
            w2sum += w2;
        } else {
            loc[i] = 0.0f;
        }
    }
    w2sum = blk_sum(w2sum, red);
    const float inv2 = 1.0f / w2sum;
#pragma unroll
    for (int i = 0; i < 8; ++i) {
        const int k = threadIdx.x + i * 256;
        if (k < n) obj[(long long)q * n + k] = (__bf16)(loc[i] * inv2);
    }
}

// ---------------------------------------------------------------------------
// Per-(row,head) L2 normalization (D=32 -> one wave32 per head-row).
// ---------------------------------------------------------------------------
__global__ __launch_bounds__(256) void l2norm_heads(
    const float* __restrict__ in, __bf16* __restrict__ out_norm,
    __bf16* __restrict__ out_raw, int n, int c, int d) {
    const int warp = (blockIdx.x * blockDim.x + threadIdx.x) >> 5;
    const int lane = threadIdx.x & 31;
    const int hh = c / d;
    if (warp >= n * hh) return;
    const int row = warp / hh;
    const int h = warp % hh;
    const long long idx = (long long)row * c + h * d + lane;
    const float v = in[idx];
    float ss = v * v;
#pragma unroll
    for (int m = 16; m >= 1; m >>= 1) ss += __shfl_xor(ss, m, 32);
    const float rn = rsqrtf(ss);
    out_norm[idx] = (__bf16)(v * rn);
    if (out_raw) out_raw[idx] = (__bf16)v;
}

// ---------------------------------------------------------------------------
// SE gate: out = raw*sig(o0) + enh*sig(o1), o = relu([raw,enh]@w1)@w2
// ---------------------------------------------------------------------------
__global__ __launch_bounds__(256) void se_fuse(
    const float* __restrict__ raw, const float* __restrict__ enh,
    const float* __restrict__ w1, const float* __restrict__ w2,
    __bf16* __restrict__ out, int total) {
    __shared__ float sw1[64], sw2[64];
    if (threadIdx.x < 64) {
        sw1[threadIdx.x] = w1[threadIdx.x];
        sw2[threadIdx.x] = w2[threadIdx.x];
    }
    __syncthreads();
    const int i = blockIdx.x * blockDim.x + threadIdx.x;
    if (i >= total) return;
    const float r = raw[i];
    const float e = enh[i];
    float o0 = 0.0f, o1 = 0.0f;
#pragma unroll
    for (int j = 0; j < 32; ++j) {
        const float hd = fmaxf(r * sw1[j] + e * sw1[32 + j], 0.0f);
        o0 += hd * sw2[2 * j];
        o1 += hd * sw2[2 * j + 1];
    }
    const float g0 = 1.0f / (1.0f + __expf(-o0));
    const float g1 = 1.0f / (1.0f + __expf(-o1));
    out[i] = (__bf16)(r * g0 + e * g1);
}

__global__ __launch_bounds__(256) void cvt_f32_bf16(
    const float* __restrict__ in, __bf16* __restrict__ out, int total) {
    const int i = blockIdx.x * blockDim.x + threadIdx.x;
    if (i < total) out[i] = (__bf16)in[i];
}

__global__ __launch_bounds__(256) void avg_bf16(
    __bf16* __restrict__ a, const __bf16* __restrict__ b, long long total) {
    const long long i = (long long)blockIdx.x * blockDim.x + threadIdx.x;
    if (i < total) a[i] = (__bf16)(0.5f * ((float)a[i] + (float)b[i]));
}

// xcat[row, 0:256] = xattn ; xcat[row, 256:512] = v_ori (f32 source)
__global__ __launch_bounds__(256) void concat_halves(
    const float* __restrict__ xattn, const float* __restrict__ vori,
    __bf16* __restrict__ out, int n, int c) {
    const int i = blockIdx.x * blockDim.x + threadIdx.x;
    const int total = n * 2 * c;
    if (i >= total) return;
    const int row = i / (2 * c);
    const int col = i % (2 * c);
    const float v = (col < c) ? xattn[(long long)row * c + col]
                              : vori[(long long)row * c + (col - c)];
    out[i] = (__bf16)v;
}

// ---------------------------------------------------------------------------
// Host-side launch sequencing
// ---------------------------------------------------------------------------
static inline void run_gemm(bool nt, const __bf16* A, const __bf16* B, float* C,
                            const float* bias, int M, int N, int K, int lda,
                            int ldb, int ldc, long long sAz, long long sBz,
                            long long sCz, int batch, float alpha,
                            hipStream_t s) {
    dim3 g((N + 63) / 64, (M + 127) / 128, batch), b(256);
    if (nt)
        gemm_bf16<true><<<g, b, 0, s>>>(A, B, C, bias, M, N, K, lda, ldb, ldc,
                                        sAz, sBz, sCz, alpha);
    else
        gemm_bf16<false><<<g, b, 0, s>>>(A, B, C, bias, M, N, K, lda, ldb, ldc,
                                         sAz, sBz, sCz, alpha);
}

extern "C" void kernel_launch(void* const* d_in, const int* in_sizes, int n_in,
                              void* d_out, int out_size, void* d_ws,
                              size_t ws_size, hipStream_t stream) {
    (void)in_sizes; (void)n_in; (void)out_size; (void)ws_size;
    const int N = N_TOK, C = C_DIM, C2 = C2_DIM, H = H_NUM;
    const long long NN = (long long)N * N;
    const long long HNN = (long long)H * NN;

    const float* x_cls = (const float*)d_in[0];
    const float* x_reg = (const float*)d_in[1];
    const float* x_key = (const float*)d_in[2];
    const float* x_edge = (const float*)d_in[3];
    const float* cls_score = (const float*)d_in[4];
    const float* fg_score = (const float*)d_in[5];
    const float* Wq_cls = (const float*)d_in[6];
    const float* Wk_cls = (const float*)d_in[7];
    const float* Wv_cls = (const float*)d_in[8];
    const float* Wq_reg = (const float*)d_in[9];
    const float* Wk_reg = (const float*)d_in[10];
    const float* Wv_reg = (const float*)d_in[11];
    const float* se_key_w1 = (const float*)d_in[12];
    const float* se_key_w2 = (const float*)d_in[13];
    const float* se_edge_w1 = (const float*)d_in[14];
    const float* se_edge_w2 = (const float*)d_in[15];
    const float* W_lin = (const float*)d_in[16];
    const float* b_lin = (const float*)d_in[17];
    const float* W_lin_reg = (const float*)d_in[18];
    const float* b_lin_reg = (const float*)d_in[19];
    float* out = (float*)d_out;

    // bump allocator over workspace
    char* wsp = (char*)d_ws;
    auto alloc = [&](size_t bytes) -> void* {
        void* p = (void*)wsp;
        wsp += (bytes + 255) & ~(size_t)255;
        return p;
    };
    const size_t NCb = (size_t)N * C * sizeof(__bf16);
    const size_t NCf = (size_t)N * C * sizeof(float);

    __bf16* xcls_bf = (__bf16*)alloc(NCb);
    __bf16* xreg_bf = (__bf16*)alloc(NCb);
    __bf16* ecls_bf = (__bf16*)alloc(NCb);
    __bf16* ereg_bf = (__bf16*)alloc(NCb);
    __bf16* Wb[6];
    for (int i = 0; i < 6; ++i) Wb[i] = (__bf16*)alloc((size_t)C * C * sizeof(__bf16));
    __bf16* Wlin_b = (__bf16*)alloc((size_t)C2 * C2 * sizeof(__bf16));
    __bf16* WlinR_b = (__bf16*)alloc((size_t)C2 * C2 * sizeof(__bf16));
    __bf16* qcn = (__bf16*)alloc(NCb);
    __bf16* kcn = (__bf16*)alloc(NCb);
    __bf16* vcn = (__bf16*)alloc(NCb);
    __bf16* qrn = (__bf16*)alloc(NCb);
    __bf16* krn = (__bf16*)alloc(NCb);
    __bf16* vrn = (__bf16*)alloc(NCb);
    __bf16* vcb = (__bf16*)alloc(NCb);
    __bf16* vrb = (__bf16*)alloc(NCb);
    __bf16* attnA = (__bf16*)alloc((size_t)HNN * sizeof(__bf16));
    __bf16* attnB = (__bf16*)alloc((size_t)HNN * sizeof(__bf16));
    __bf16* sim2 = (__bf16*)alloc((size_t)NN * sizeof(__bf16));
    __bf16* obj = (__bf16*)alloc((size_t)NN * sizeof(__bf16));
    __bf16* xcat_c = (__bf16*)alloc((size_t)N * C2 * sizeof(__bf16));
    __bf16* xcat_r = (__bf16*)alloc((size_t)N * C2 * sizeof(__bf16));
    float* qc = (float*)alloc(NCf);
    float* kc = (float*)alloc(NCf);
    float* vc = (float*)alloc(NCf);
    float* qr = (float*)alloc(NCf);
    float* kr = (float*)alloc(NCf);
    float* vr = (float*)alloc(NCf);
    float* logits = (float*)alloc((size_t)HNN * sizeof(float));
    float* acr = (float*)alloc((size_t)NN * sizeof(float));
    float* arr = (float*)alloc((size_t)NN * sizeof(float));
    float* xat_c = (float*)alloc(NCf);
    float* xat_r = (float*)alloc(NCf);

    const int NC = N * C;
    auto cgrid = [](int total) { return dim3((total + 255) / 256); };

    // 1) precision conversion of activations & weights
    cvt_f32_bf16<<<cgrid(NC), 256, 0, stream>>>(x_cls, xcls_bf, NC);
    cvt_f32_bf16<<<cgrid(NC), 256, 0, stream>>>(x_reg, xreg_bf, NC);
    const float* Wsrc[6] = {Wq_cls, Wk_cls, Wv_cls, Wq_reg, Wk_reg, Wv_reg};
    for (int i = 0; i < 6; ++i)
        cvt_f32_bf16<<<cgrid(C * C), 256, 0, stream>>>(Wsrc[i], Wb[i], C * C);
    cvt_f32_bf16<<<cgrid(C2 * C2), 256, 0, stream>>>(W_lin, Wlin_b, C2 * C2);
    cvt_f32_bf16<<<cgrid(C2 * C2), 256, 0, stream>>>(W_lin_reg, WlinR_b, C2 * C2);

    // 2) SE fusion (N1==N2 => local == enh, compute once per branch)
    se_fuse<<<cgrid(NC), 256, 0, stream>>>(x_cls, x_key, se_key_w1, se_key_w2, ecls_bf, NC);
    se_fuse<<<cgrid(NC), 256, 0, stream>>>(x_reg, x_edge, se_edge_w1, se_edge_w2, ereg_bf, NC);

    // 3) projections (NN GEMMs, M=2048 N=256 K=256)
    run_gemm(false, ecls_bf, Wb[0], qc, nullptr, N, C, C, C, C, C, 0, 0, 0, 1, 1.0f, stream);
    run_gemm(false, ecls_bf, Wb[1], kc, nullptr, N, C, C, C, C, C, 0, 0, 0, 1, 1.0f, stream);
    run_gemm(false, xcls_bf, Wb[2], vc, nullptr, N, C, C, C, C, C, 0, 0, 0, 1, 1.0f, stream);
    run_gemm(false, ereg_bf, Wb[3], qr, nullptr, N, C, C, C, C, C, 0, 0, 0, 1, 1.0f, stream);
    run_gemm(false, ereg_bf, Wb[4], kr, nullptr, N, C, C, C, C, C, 0, 0, 0, 1, 1.0f, stream);
    run_gemm(false, xreg_bf, Wb[5], vr, nullptr, N, C, C, C, C, C, 0, 0, 0, 1, 1.0f, stream);

    // 4) per-head L2 normalization -> bf16 (v also kept raw in bf16)
    const dim3 ng((N * H + 7) / 8);
    l2norm_heads<<<ng, 256, 0, stream>>>(qc, qcn, nullptr, N, C, D_DIM);
    l2norm_heads<<<ng, 256, 0, stream>>>(kc, kcn, nullptr, N, C, D_DIM);
    l2norm_heads<<<ng, 256, 0, stream>>>(vc, vcn, vcb, N, C, D_DIM);
    l2norm_heads<<<ng, 256, 0, stream>>>(qr, qrn, nullptr, N, C, D_DIM);
    l2norm_heads<<<ng, 256, 0, stream>>>(kr, krn, nullptr, N, C, D_DIM);
    l2norm_heads<<<ng, 256, 0, stream>>>(vr, vrn, vrb, N, C, D_DIM);

    // 5) cls attention: per-head qk^T (NT, K=32, batched over 8 heads)
    run_gemm(true, qcn, kcn, logits, nullptr, N, N, D_DIM, C, C, N, D_DIM, D_DIM, NN, H, 1.0f, stream);
    softmax_rows<<<dim3(N, H), 256, 0, stream>>>(logits, attnA, cls_score, N, SCALE_F);

    // 6) reg attention (reuse logits scratch)
    run_gemm(true, qrn, krn, logits, nullptr, N, N, D_DIM, C, C, N, D_DIM, D_DIM, NN, H, 1.0f, stream);
    softmax_rows<<<dim3(N, H), 256, 0, stream>>>(logits, attnB, nullptr, N, SCALE_F);

    // 7) raw similarity grams: mean_h(per-head gram) == (vn @ vn^T)/H
    run_gemm(true, vcn, vcn, acr, nullptr, N, N, C, C, C, N, 0, 0, 0, 1, 1.0f / H, stream);
    run_gemm(true, vrn, vrn, arr, nullptr, N, N, C, C, C, N, 0, 0, 0, 1, 1.0f / H, stream);

    // 8) attn = 0.5*(cls + reg), in place into attnA
    avg_bf16<<<dim3((unsigned)((HNN + 255) / 256)), 256, 0, stream>>>(attnA, attnB, HNN);

    // 9) second-round similarity gating masks
    sim_round2_kernel<<<dim3(N), 256, 0, stream>>>(attnA, acr, arr, cls_score,
                                                   fg_score, sim2, obj, N);

    // 10) AV: per-head attn @ v (NN, K=2048, N=32, batched over heads)
    run_gemm(false, attnA, vcb, xat_c, nullptr, N, D_DIM, N, N, C, C, NN, D_DIM, D_DIM, H, 1.0f, stream);
    run_gemm(false, attnA, vrb, xat_r, nullptr, N, D_DIM, N, N, C, C, NN, D_DIM, D_DIM, H, 1.0f, stream);

    // 11) concat [x, x_ori] -> bf16
    concat_halves<<<cgrid(N * C2), 256, 0, stream>>>(xat_c, vc, xcat_c, N, C);
    concat_halves<<<cgrid(N * C2), 256, 0, stream>>>(xat_r, vr, xcat_r, N, C);

    // 12) output linears straight into d_out columns [256:768), ldc = 768
    const int ldo = 3 * C;  // 768
    run_gemm(false, xcat_c, Wlin_b, out + C, b_lin, N, C2, C2, C2, C2, ldo, 0, 0, 0, 1, 1.0f, stream);
    run_gemm(false, xcat_r, WlinR_b, out + (long long)N * ldo + C, b_lin_reg,
             N, C2, C2, C2, C2, ldo, 0, 0, 0, 1, 1.0f, stream);

    // 13) second-round aggregation into d_out columns [0:256)
    run_gemm(false, sim2, vcb, out, nullptr, N, C, N, N, C, ldo, 0, 0, 0, 1, 1.0f, stream);
    run_gemm(false, obj, vrb, out + (long long)N * ldo, nullptr, N, C, N, N, C, ldo, 0, 0, 0, 1, 1.0f, stream);
}
